// StdNormalSinkhornMatcher_56727928045839
// MI455X (gfx1250) — compile-verified, matrix operand order checked by reading, not run on hardware
//
#include <hip/hip_runtime.h>
#include <hip/hip_bf16.h>
#include <math.h>

typedef __attribute__((ext_vector_type(16))) __bf16 v16bf;
typedef __attribute__((ext_vector_type(8)))  float  v8f;

#define DK     512
#define EPS_F  0.05f
#define N_IT   50

// ---------------------------------------------------------------- helpers
__device__ __forceinline__ unsigned short f32_to_bf16_rne(float f) {
  unsigned int u = __float_as_uint(f);
  unsigned int r = u + 0x7FFFu + ((u >> 16) & 1u);
  if ((u & 0x7F800000u) == 0x7F800000u) r = u;   // inf/nan passthrough
  return (unsigned short)(r >> 16);
}

// ------------------------------------------------ bf16 convert + sq-norms
// one block (128 threads) per row of 512 floats
__global__ void prep_kernel(const float* __restrict__ src,
                            unsigned short* __restrict__ dst,
                            float* __restrict__ norms) {
  int row = blockIdx.x;
  int tid = threadIdx.x;
  const float* s = src + (size_t)row * DK;
  unsigned short* d = dst + (size_t)row * DK;
  float4 v = *reinterpret_cast<const float4*>(s + tid * 4);
  unsigned int lo = (unsigned int)f32_to_bf16_rne(v.x) |
                    ((unsigned int)f32_to_bf16_rne(v.y) << 16);
  unsigned int hi = (unsigned int)f32_to_bf16_rne(v.z) |
                    ((unsigned int)f32_to_bf16_rne(v.w) << 16);
  *reinterpret_cast<uint2*>(d + tid * 4) = make_uint2(lo, hi);
  float sq = v.x * v.x + v.y * v.y + v.z * v.z + v.w * v.w;
  for (int o = 16; o; o >>= 1) sq += __shfl_xor(sq, o, 32);
  __shared__ float wsum[4];
  if ((tid & 31) == 0) wsum[tid >> 5] = sq;
  __syncthreads();
  if (tid == 0) norms[row] = wsum[0] + wsum[1] + wsum[2] + wsum[3];
}

// ------------------------------------------------ WMMA distance GEMM
// C[i,j] = sqrt(max(nA[i] + nB[j] - 2 * A_i . B_j, 1e-12))
#define BM 128
#define BN 128
#define BK 32
#define LST 40   // LDS row stride in bf16 elements (80 B, 16B-aligned, conflict-free)

union FragU { uint4 q[2]; v16bf v; };

__global__ __launch_bounds__(256)
void dist_gemm_kernel(const unsigned short* __restrict__ A,
                      const unsigned short* __restrict__ Bm,
                      const float* __restrict__ nA,
                      const float* __restrict__ nB,
                      float* __restrict__ C, int N) {
  __shared__ __align__(16) unsigned short As[BM * LST];
  __shared__ __align__(16) unsigned short Bs[BN * LST];

  int tid  = threadIdx.x;
  int lane = tid & 31, w = tid >> 5;
  int wm = w & 3;                 // 4 waves along M, 32 rows each
  int wn = w >> 2;                // 2 waves along N, 64 cols each
  int lane16 = lane & 15, hi = lane >> 4;
  size_t baseM = (size_t)blockIdx.y * BM;
  size_t baseN = (size_t)blockIdx.x * BN;

  v8f acc[2][4] = {};

  int lr = tid >> 1;              // 0..127 : tile row
  int lc = (tid & 1) * 16;        // 0 / 16 : first K element this thread stages

  const unsigned short* Ag = A  + (baseM + (size_t)lr) * DK + lc;
  const unsigned short* Bg = Bm + (baseN + (size_t)lr) * DK + lc;
  unsigned short* Al = As + lr * LST + lc;
  unsigned short* Bl = Bs + lr * LST + lc;

  for (int k0 = 0; k0 < DK; k0 += BK) {
    __syncthreads();
    uint4 a0 = *reinterpret_cast<const uint4*>(Ag + k0);
    uint4 a1 = *reinterpret_cast<const uint4*>(Ag + k0 + 8);
    uint4 b0 = *reinterpret_cast<const uint4*>(Bg + k0);
    uint4 b1 = *reinterpret_cast<const uint4*>(Bg + k0 + 8);
    *reinterpret_cast<uint4*>(Al)     = a0;
    *reinterpret_cast<uint4*>(Al + 8) = a1;
    *reinterpret_cast<uint4*>(Bl)     = b0;
    *reinterpret_cast<uint4*>(Bl + 8) = b1;
    __syncthreads();

    FragU af[2], bf[4];
    // 16-bit A layout: lanes 0-15 K=0..7 & 16..23, lanes 16-31 K=8..15 & 24..31
    for (int mi = 0; mi < 2; ++mi) {
      const unsigned short* p = As + (wm * 32 + mi * 16 + lane16) * LST + hi * 8;
      af[mi].q[0] = *reinterpret_cast<const uint4*>(p);
      af[mi].q[1] = *reinterpret_cast<const uint4*>(p + 16);
    }
    for (int ni = 0; ni < 4; ++ni) {
      const unsigned short* p = Bs + (wn * 64 + ni * 16 + lane16) * LST + hi * 8;
      bf[ni].q[0] = *reinterpret_cast<const uint4*>(p);
      bf[ni].q[1] = *reinterpret_cast<const uint4*>(p + 16);
    }
    for (int mi = 0; mi < 2; ++mi)
      for (int ni = 0; ni < 4; ++ni)
        acc[mi][ni] = __builtin_amdgcn_wmma_f32_16x16x32_bf16(
            false, af[mi].v, false, bf[ni].v, (short)0, acc[mi][ni],
            false, false);
  }

  // C/D layout: lane 0-15 -> N=lane, M=r ; lane 16-31 -> N=lane-16, M=r+8
  for (int mi = 0; mi < 2; ++mi) {
    for (int ni = 0; ni < 4; ++ni) {
      int col = (int)baseN + wn * 64 + ni * 16 + lane16;
      float nb = nB[col];
      for (int r = 0; r < 8; ++r) {
        int row = (int)baseM + wm * 32 + mi * 16 + r + hi * 8;
        float val = nA[row] + nb - 2.0f * acc[mi][ni][r];
        C[(size_t)row * (size_t)N + col] = sqrtf(fmaxf(val, 1e-12f));
      }
    }
  }
}

// ------------------------------------------------ fused Sinkhorn step
// segment 0: f' from (g - Cxy)/eps row-lse     segment 1: g' from (f - Cyx)/eps
// segment 2: p' from (p - Cxx)/eps             segment 3: q' from (q - Cyy)/eps
__global__ __launch_bounds__(256)
void lse_step_kernel(const float* __restrict__ Cxy, const float* __restrict__ Cyx,
                     const float* __restrict__ Cxx, const float* __restrict__ Cyy,
                     const float* __restrict__ f,  const float* __restrict__ g,
                     const float* __restrict__ p,  const float* __restrict__ q,
                     float* __restrict__ fo, float* __restrict__ go,
                     float* __restrict__ po, float* __restrict__ qo,
                     int n, float logc, int damped) {
  extern __shared__ float sPot[];
  int bps = gridDim.x >> 2;
  int seg = blockIdx.x / bps;
  int blk = blockIdx.x - seg * bps;

  const float *M, *pot, *oldv; float* outv;
  switch (seg) {
    case 0:  M = Cxy; pot = g; oldv = f; outv = fo; break;
    case 1:  M = Cyx; pot = f; oldv = g; outv = go; break;
    case 2:  M = Cxx; pot = p; oldv = p; outv = po; break;
    default: M = Cyy; pot = q; oldv = q; outv = qo; break;
  }

  int tid = threadIdx.x;
  for (int i = tid * 4; i < n; i += 256 * 4)
    *reinterpret_cast<float4*>(sPot + i) =
        *reinterpret_cast<const float4*>(pot + i);
  __syncthreads();

  int lane = tid & 31, w = tid >> 5;
  int row = blk * 8 + w;                       // one wave per row
  const float* Mr = M + (size_t)row * (size_t)n;
  const float inv_eps = 1.0f / EPS_F;

  float m = -INFINITY, s = 0.0f;
  for (int j0 = lane * 4; j0 < n; j0 += 128) {
    float4 c  = *reinterpret_cast<const float4*>(Mr + j0);
    float4 gv = *reinterpret_cast<const float4*>(sPot + j0);
    float v0 = (gv.x - c.x) * inv_eps;
    float v1 = (gv.y - c.y) * inv_eps;
    float v2 = (gv.z - c.z) * inv_eps;
    float v3 = (gv.w - c.w) * inv_eps;
    float cm = fmaxf(fmaxf(v0, v1), fmaxf(v2, v3));
    float nm = fmaxf(m, cm);
    s = s * __expf(m - nm) + __expf(v0 - nm) + __expf(v1 - nm) +
        __expf(v2 - nm) + __expf(v3 - nm);
    m = nm;
  }
  for (int o = 16; o; o >>= 1) {
    float om = __shfl_xor(m, o, 32);
    float os = __shfl_xor(s, o, 32);
    float nm = fmaxf(m, om);
    s = s * __expf(m - nm) + os * __expf(om - nm);
    m = nm;
  }
  if (lane == 0) {
    float lse = m + __logf(s) + logc;
    outv[row] = damped ? 0.5f * (oldv[row] - EPS_F * lse) : (-EPS_F * lse);
  }
}

// ------------------------------------------------ small kernels
__global__ void fill_zero_kernel(float* ptr, int count) {
  int i = blockIdx.x * blockDim.x + threadIdx.x;
  if (i < count) ptr[i] = 0.0f;
}

__global__ void final_dot_kernel(const float* __restrict__ fe,
                                 const float* __restrict__ ge,
                                 const float* __restrict__ pe,
                                 const float* __restrict__ qe,
                                 float* __restrict__ out, int n) {
  __shared__ float red[256];
  float acc = 0.0f;
  for (int i = threadIdx.x; i < n; i += 256)
    acc += (fe[i] - pe[i]) + (ge[i] - qe[i]);
  red[threadIdx.x] = acc;
  __syncthreads();
  for (int st = 128; st; st >>= 1) {
    if (threadIdx.x < st) red[threadIdx.x] += red[threadIdx.x + st];
    __syncthreads();
  }
  if (threadIdx.x == 0) out[0] = red[0] / (float)n;
}

// ------------------------------------------------ launch
extern "C" void kernel_launch(void* const* d_in, const int* in_sizes, int n_in,
                              void* d_out, int out_size, void* d_ws, size_t ws_size,
                              hipStream_t stream) {
  const float* x  = (const float*)d_in[0];   // "x"
  const float* pz = (const float*)d_in[1];   // "prior_z"   (xs = prior_z, xt = x)
  int B = in_sizes[0] / DK;                  // 8192
  size_t n2 = (size_t)B * (size_t)B;

  float* ws  = (float*)d_ws;
  float* Cxy = ws;                 // dist(pz, x)
  float* Cyx = Cxy + n2;           // dist(x, pz) = Cxy^T, contiguous for col-lse
  float* Cxx = Cyx + n2;           // dist(pz, pz)
  float* Cyy = Cxx + n2;           // dist(x, x)
  unsigned short* Pb = (unsigned short*)(Cyy + n2);
  unsigned short* Xb = Pb + (size_t)B * DK;
  float* nP   = (float*)(Xb + (size_t)B * DK);
  float* nX   = nP + B;
  float* potA = nX + B;            // f,g,p,q (4*B)
  float* potB = potA + 4 * (size_t)B;

  float logc = -logf((float)B);    // loga == logb == -log n

  prep_kernel<<<B, 128, 0, stream>>>(pz, Pb, nP);
  prep_kernel<<<B, 128, 0, stream>>>(x,  Xb, nX);

  dim3 gg(B / BN, B / BM);
  dist_gemm_kernel<<<gg, 256, 0, stream>>>(Pb, Xb, nP, nX, Cxy, B);
  dist_gemm_kernel<<<gg, 256, 0, stream>>>(Xb, Pb, nX, nP, Cyx, B);
  dist_gemm_kernel<<<gg, 256, 0, stream>>>(Pb, Pb, nP, nP, Cxx, B);
  dist_gemm_kernel<<<gg, 256, 0, stream>>>(Xb, Xb, nX, nX, Cyy, B);

  fill_zero_kernel<<<(4 * B + 255) / 256, 256, 0, stream>>>(potA, 4 * B);

  int bps = B / 8;
  dim3 lg(4 * bps);
  size_t shmem = (size_t)B * sizeof(float);
  float* cur = potA;
  float* nxt = potB;
  for (int it = 0; it < N_IT; ++it) {
    lse_step_kernel<<<lg, 256, shmem, stream>>>(
        Cxy, Cyx, Cxx, Cyy,
        cur, cur + B, cur + 2 * B, cur + 3 * B,
        nxt, nxt + B, nxt + 2 * B, nxt + 3 * B, B, logc, 1);
    float* t = cur; cur = nxt; nxt = t;
  }
  // one undamped extrapolation step -> f_e, g_e, p_e, q_e
  lse_step_kernel<<<lg, 256, shmem, stream>>>(
      Cxy, Cyx, Cxx, Cyy,
      cur, cur + B, cur + 2 * B, cur + 3 * B,
      nxt, nxt + B, nxt + 2 * B, nxt + 3 * B, B, logc, 0);

  final_dot_kernel<<<1, 256, 0, stream>>>(nxt, nxt + B, nxt + 2 * B, nxt + 3 * B,
                                          (float*)d_out, B);
}
